// TetrahedralCore_15152644620951
// MI455X (gfx1250) — matrix operand order repaired and checked by reading, not compiled, not run on hardware
//
#include <hip/hip_runtime.h>

// ---------------------------------------------------------------------------
// TetrahedralCore collapses to a single affine map:  out = x @ Weff + beff.
// (softmax over a size-1 axis == 1.0, so attention is identity and q/k are
// dead; every remaining op is linear.)  We fold all weights on-device into
// Weff (512x512) / beff (512) in fp32, convert Weff to fp16 (transposed for
// WMMA-B loading), then run one memory-bound GEMM with v_wmma_f32_16x16x32_f16.
// ---------------------------------------------------------------------------

typedef __attribute__((ext_vector_type(16))) _Float16 v16h;
typedef __attribute__((ext_vector_type(8)))  float    v8f;

union FragU { v16h v; int4 i[2]; };

__device__ __forceinline__ unsigned short f2h(float f) {
    _Float16 h = (_Float16)f;
    return __builtin_bit_cast(unsigned short, h);
}

// ------------------------- workspace layout (floats) ------------------------
#define WS_E      0          // 256*64
#define WS_G      16384      // 256*64
#define WS_E0     32768      // 64
#define WS_G0     32832      // 64
#define WS_A1     32896      // 512*256
#define WS_A2     163968     // 512*256
#define WS_B1     295040     // 256
#define WS_B2     295296     // 256
#define WS_BEFF   295552     // 512
#define WS_WT_F   296064     // ushort WeffT[512*512] lives here (byte off 1184256)
#define WS_BYTES_NEEDED (1184256 + 512*512*2)

// --- E (edge-mean collapse): E[u*64+i][j], e0[j] ---------------------------
__global__ void k_build_E(const float* __restrict__ W_edge,
                          const float* __restrict__ b_edge,
                          float* __restrict__ E, float* __restrict__ e0) {
    int id = blockIdx.x * blockDim.x + threadIdx.x;
    const int ev[4][3] = {{0,1,2},{3,4,0},{5,1,3},{2,4,5}};
    const int eh[4][3] = {{0,0,0},{0,0,1},{0,1,1},{1,1,1}};
    if (id < 16384) {
        int row = id >> 6, j = id & 63;
        int u = row >> 6, i = row & 63;
        float s = 0.f;
        #pragma unroll
        for (int t = 0; t < 3; t++) {
            int e = ev[u][t], h = eh[u][t];
            s += W_edge[e * 8192 + (h * 64 + i) * 64 + j];
        }
        E[id] = s * (1.f / 6.f);
    } else if (id < 16448) {
        int j = id - 16384;
        float s = 0.f;
        #pragma unroll
        for (int e = 0; e < 6; e++) s += b_edge[e * 64 + j];
        e0[j] = s * (1.f / 6.f);
    }
}

// --- G (face-mean collapse incl. attended=val path): G[u*64+i][j], g0[j] ---
__global__ void k_build_G(const float* __restrict__ W_qkv,
                          const float* __restrict__ W_faceout,
                          const float* __restrict__ b_qkv,
                          const float* __restrict__ b_faceout,
                          float* __restrict__ G, float* __restrict__ g0) {
    int id = blockIdx.x * blockDim.x + threadIdx.x;
    const int gf[4][3] = {{0,1,2},{0,1,3},{0,2,3},{1,2,3}};
    const int gs[4][3] = {{0,0,0},{1,1,0},{2,1,1},{2,2,2}};
    if (id < 16384) {
        int row = id >> 6, j = id & 63;
        int u = row >> 6, i = row & 63;
        float s = 0.f;
        #pragma unroll
        for (int t = 0; t < 3; t++) {
            int f = gf[u][t], sl = gs[u][t];
            const float* q  = W_qkv + f * 36864 + (sl * 64 + i) * 192 + 128;
            const float* wf = W_faceout + f * 4096;
            float d = 0.f;
            for (int k = 0; k < 64; k++) d += q[k] * wf[k * 64 + j];
            s += d + (1.f / 3.f) * wf[i * 64 + j];
        }
        G[id] = 0.25f * s;
    } else if (id < 16448) {
        int j = id - 16384;
        float s = 0.f;
        #pragma unroll
        for (int f = 0; f < 4; f++) {
            const float* wf = W_faceout + f * 4096;
            float d = 0.f;
            for (int k = 0; k < 64; k++) d += b_qkv[f * 192 + 128 + k] * wf[k * 64 + j];
            s += d + b_faceout[f * 64 + j];
        }
        g0[j] = 0.25f * s;
    }
}

// --- affine residual step: Y[r][c] = X[r][c] + sum_k X[r][k]*M[k][c&63] (+cvec)
__global__ void k_affine_step(const float* __restrict__ X, const float* __restrict__ M,
                              const float* __restrict__ cvec, float* __restrict__ Y,
                              int R) {
    int id = blockIdx.x * blockDim.x + threadIdx.x;
    if (id >= R * 256) return;
    int r = id >> 8, c = id & 255, j = c & 63;
    const float* xr = X + r * 256;
    float acc = 0.f;
    #pragma unroll 4
    for (int k = 0; k < 256; k++) acc += xr[k] * M[k * 64 + j];
    float y = xr[c] + acc;
    if (cvec) y += cvec[j];
    Y[id] = y;
}

// --- Weff = A2 @ W_out, stored transposed as fp16: WeffT[n][k] ------------
__global__ void k_compose(const float* __restrict__ A2, const float* __restrict__ W_out,
                          unsigned short* __restrict__ WT) {
    int id = blockIdx.x * blockDim.x + threadIdx.x;   // 262144 threads
    int n = id & 511, k = id >> 9;
    const float* ar = A2 + k * 256;
    float acc = 0.f;
    #pragma unroll 4
    for (int c = 0; c < 256; c++) acc += ar[c] * W_out[c * 512 + n];
    WT[n * 512 + k] = f2h(acc);
}

__global__ void k_bias_out(const float* __restrict__ b2, const float* __restrict__ W_out,
                           const float* __restrict__ b_out, float* __restrict__ beff) {
    int n = blockIdx.x * blockDim.x + threadIdx.x;
    if (n >= 512) return;
    float acc = b_out[n];
    for (int c = 0; c < 256; c++) acc += b2[c] * W_out[c * 512 + n];
    beff[n] = acc;
}

// --------------------------- main fused GEMM -------------------------------
// out(65536x512) = x(65536x512,f32) @ Weff(512x512,f16 via WeffT) + beff
// Block: 256 thr (8 wave32), tile M=32 x N=512 (full width -> x read once).
// Waves 2(M) x 4(N); wave tile 16x128 = 8 WMMA accumulators (64 VGPR).
__global__ __launch_bounds__(256) void k_gemm(
    const float* __restrict__ X, const unsigned short* __restrict__ WT,
    const float* __restrict__ beff, float* __restrict__ out) {

    __shared__ __align__(16) unsigned short As[32 * 32];   //  2 KB, [row][k]
    __shared__ __align__(16) unsigned short Bs[512 * 32];  // 32 KB, [n][k]

    const int tid  = threadIdx.x;
    const int lane = tid & 31;
    const int wave = tid >> 5;
    const int wm   = wave & 1;        // M half
    const int wq   = wave >> 1;       // N quarter
    const int lm   = lane & 15;
    const int hi   = lane >> 4;
    const long mBase = (long)blockIdx.x * 32;

    v8f acc[8];
    #pragma unroll
    for (int ns = 0; ns < 8; ns++) acc[ns] = (v8f){0.f,0.f,0.f,0.f,0.f,0.f,0.f,0.f};

    for (int k0 = 0; k0 < 512; k0 += 32) {
        // stage A: 32x32 f32 -> fp16 LDS (coalesced float4 loads)
        {
            int row = tid >> 3, c4 = tid & 7;
            const float* gp = X + (mBase + row) * 512 + k0 + c4 * 4;
            float4 v = *(const float4*)gp;
            if (k0 + 32 < 512) __builtin_prefetch(gp + 32, 0, 1);
            uint2 p;
            p.x = ((unsigned)f2h(v.y) << 16) | f2h(v.x);
            p.y = ((unsigned)f2h(v.w) << 16) | f2h(v.z);
            *(uint2*)&As[row * 32 + c4 * 4] = p;
        }
        // stage B: 512x32 f16 from L2-resident WeffT (int4 copies)
        #pragma unroll
        for (int i = 0; i < 8; i++) {
            int idx = tid + i * 256;          // 0..2047
            int n = idx >> 2, c = idx & 3;
            *(int4*)&Bs[n * 32 + c * 8] =
                *(const int4*)(WT + (long)n * 512 + k0 + c * 8);
        }
        __syncthreads();

        // A fragment: lane<16 -> K{0..7,16..23}, lane>=16 -> K{8..15,24..31}
        FragU a;
        const unsigned short* ap = &As[(wm * 16 + lm) * 32 + hi * 8];
        a.i[0] = *(const int4*)ap;
        a.i[1] = *(const int4*)(ap + 16);

        #pragma unroll
        for (int ns = 0; ns < 8; ns++) {
            // B fragment: col = lane&15, lane<16 -> K{0..7,16..23}, lane>=16 -> K{8..15,24..31}
            FragU b;
            int col = wq * 128 + ns * 16 + lm;
            const unsigned short* bp = &Bs[col * 32 + hi * 8];
            b.i[0] = *(const int4*)bp;
            b.i[1] = *(const int4*)(bp + 16);
            acc[ns] = __builtin_amdgcn_wmma_f32_16x16x32_f16(
                false, a.v, false, b.v, (short)0, acc[ns], false, false);
        }
        __syncthreads();
    }

    // epilogue: D layout -> row = mBase + wm*16 + hi*8 + v, col = base + lm
    const long rowBase = mBase + wm * 16 + hi * 8;
    #pragma unroll
    for (int ns = 0; ns < 8; ns++) {
        int col = wq * 128 + ns * 16 + lm;
        float bv = beff[col];
        float* op = out + rowBase * 512 + col;
        #pragma unroll
        for (int v = 0; v < 8; v++) op[v * 512] = acc[ns][v] + bv;
    }
}

// ---------------------------------------------------------------------------
extern "C" void kernel_launch(void* const* d_in, const int* in_sizes, int n_in,
                              void* d_out, int out_size, void* d_ws, size_t ws_size,
                              hipStream_t stream) {
    const float* x         = (const float*)d_in[0];
    const float* W_embed   = (const float*)d_in[1];
    const float* b_embed   = (const float*)d_in[2];
    const float* W_edge    = (const float*)d_in[3];
    const float* b_edge    = (const float*)d_in[4];
    const float* W_qkv     = (const float*)d_in[5];
    const float* b_qkv     = (const float*)d_in[6];
    const float* W_faceout = (const float*)d_in[7];
    const float* b_faceout = (const float*)d_in[8];
    const float* W_out     = (const float*)d_in[9];
    const float* b_out     = (const float*)d_in[10];
    float* outp = (float*)d_out;

    if (ws_size < (size_t)WS_BYTES_NEEDED) return;  // leaves d_out poisoned -> loud failure

    float* ws = (float*)d_ws;
    float* E    = ws + WS_E;
    float* G    = ws + WS_G;
    float* e0   = ws + WS_E0;
    float* g0   = ws + WS_G0;
    float* A1   = ws + WS_A1;
    float* A2   = ws + WS_A2;
    float* b1   = ws + WS_B1;
    float* b2   = ws + WS_B2;
    float* beff = ws + WS_BEFF;
    unsigned short* WT = (unsigned short*)(ws + WS_WT_F);

    // fold weights into (Weff, beff)
    k_build_E<<<257, 64, 0, stream>>>(W_edge, b_edge, E, e0);
    k_build_G<<<257, 64, 0, stream>>>(W_qkv, W_faceout, b_qkv, b_faceout, G, g0);
    k_affine_step<<<512, 256, 0, stream>>>(W_embed, E, nullptr, A1, 512);
    k_affine_step<<<1, 256, 0, stream>>>(b_embed, E, e0, b1, 1);
    k_affine_step<<<512, 256, 0, stream>>>(A1, G, nullptr, A2, 512);
    k_affine_step<<<1, 256, 0, stream>>>(b1, G, g0, b2, 1);
    k_compose<<<1024, 256, 0, stream>>>(A2, W_out, WT);
    k_bias_out<<<2, 256, 0, stream>>>(b2, W_out, b_out, beff);

    // main memory-bound WMMA GEMM: 65536/32 = 2048 workgroups
    k_gemm<<<2048, 256, 0, stream>>>(x, WT, beff, outp);
}
